// LSTM_48610439856222
// MI455X (gfx1250) — matrix operand order repaired and checked by reading, not compiled, not run on hardware
//
#include <hip/hip_runtime.h>
#include <hip/hip_bf16.h>

// Problem sizes (fixed by the reference)
#define BB 64
#define SS 512
#define DD 512
#define HH 1024
#define OO 256

typedef __attribute__((ext_vector_type(16))) __bf16 v16bf;
typedef __attribute__((ext_vector_type(8)))  __bf16 v8bf;
typedef __attribute__((ext_vector_type(8)))  float  v8f;

union Frag { v16bf v; v8bf h[2]; };

// ---------- helpers ----------

// Native f32 -> bf16 (hardware RNE convert on gfx1250).
__device__ __forceinline__ __bf16 f2bf(float f) { return (__bf16)f; }

__device__ __forceinline__ v8bf ld8(const __bf16* p) { return *(const v8bf*)p; }

// A fragment (16x32 bf16, row-major source, row stride lda elements).
// ISA layout: lanes 0-15 -> M=lane, K {0..7} in v0-3, {16..23} in v4-7;
//             lanes 16-31 -> M=lane-16, K {8..15}, {24..31}.
__device__ __forceinline__ v16bf load_a_bf16(const __bf16* __restrict__ A, int lda,
                                             int m0, int k0, int lane) {
  int hf = lane >> 4, mr = lane & 15;
  const __bf16* base = A + (size_t)(m0 + mr) * lda;
  Frag f;
  f.h[0] = ld8(base + k0 + 8 * hf);
  f.h[1] = ld8(base + k0 + 16 + 8 * hf);
  return f.v;
}

__device__ __forceinline__ v8bf cvt8(const float* __restrict__ p) {
  const float4* q = (const float4*)p;
  float4 a = q[0], b = q[1];
  v8bf r;
  r[0] = f2bf(a.x); r[1] = f2bf(a.y); r[2] = f2bf(a.z); r[3] = f2bf(a.w);
  r[4] = f2bf(b.x); r[5] = f2bf(b.y); r[6] = f2bf(b.z); r[7] = f2bf(b.w);
  return r;
}

// A fragment from fp32 source (fallback path if workspace is too small for xb).
__device__ __forceinline__ v16bf load_a_f32(const float* __restrict__ A, int lda,
                                            int m0, int k0, int lane) {
  int hf = lane >> 4, mr = lane & 15;
  const float* base = A + (size_t)(m0 + mr) * lda;
  Frag f;
  f.h[0] = cvt8(base + k0 + 8 * hf);
  f.h[1] = cvt8(base + k0 + 16 + 8 * hf);
  return f.v;
}

// B fragment (32x16 bf16). B[k,n] = W[n,k], W row-major [Ntot, ldk].
// ISA layout: lanes 0-15 -> col=lane, K {0..7},{16..23}; lanes 16-31 -> col=lane-16, K {8..15},{24..31}.
__device__ __forceinline__ v16bf load_b(const __bf16* __restrict__ W, int ldk,
                                        int n0, int k0, int lane) {
  int hf = lane >> 4;
  const __bf16* base = W + (size_t)(n0 + (lane & 15)) * ldk + k0 + 8 * hf;
  Frag f;
  f.h[0] = ld8(base);
  f.h[1] = ld8(base + 16);
  return f.v;
}

__device__ __forceinline__ v8f wmma_bf16(v16bf a, v16bf b, v8f c) {
  // (neg_a, A, neg_b, B, c_mod, C, reuse_a, reuse_b)
  return __builtin_amdgcn_wmma_f32_16x16x32_bf16(false, a, false, b, (short)0, c,
                                                 false, false);
}

__device__ __forceinline__ float sigf(float x) { return 1.0f / (1.0f + __expf(-x)); }

// ---------- kernels ----------

__global__ void cvt_f32_bf16_kernel(const float* __restrict__ in,
                                    __bf16* __restrict__ out, int n) {
  int i = blockIdx.x * blockDim.x + threadIdx.x;
  if (i < n) out[i] = f2bf(in[i]);
}

// One LSTM cell step for one layer.
// Wave w owns a 16x16 (batch x hidden) block and all 4 gates for it.
// gates = inp@Wx^T + h_prev@Wh^T (+ biases), then in-register cell update.
__global__ void __launch_bounds__(256) lstm_step_kernel(
    const float*  __restrict__ xA,      // fp32 A source (fallback) or nullptr
    const __bf16* __restrict__ hA,      // bf16 A source or nullptr
    int lda, int Kx,                    // A row stride / K of the Wx GEMM
    const __bf16* __restrict__ Wx,      // [4H, Kx] bf16
    const __bf16* __restrict__ Wh,      // [4H, H]  bf16
    const __bf16* __restrict__ h_prev,  // [B, H]   bf16
    const float*  __restrict__ bx, const float* __restrict__ bh,
    float*  __restrict__ c,             // [B, H] fp32 (in-place)
    __bf16* __restrict__ h_out)         // [B, H] bf16
{
  int lane = threadIdx.x & 31;
  int wid  = (blockIdx.x * blockDim.x + threadIdx.x) >> 5;  // 0..255
  int n0 = (wid & 63) << 4;   // hidden tile (0..1008)
  int m0 = (wid >> 6) << 4;   // batch tile  (0,16,32,48)

  v8f acc[4];
#pragma unroll
  for (int g = 0; g < 4; ++g)
#pragma unroll
    for (int r = 0; r < 8; ++r) acc[g][r] = 0.0f;

  // Phase 1: inp @ Wx^T
  if (hA) {
#pragma unroll 2
    for (int k0 = 0; k0 < Kx; k0 += 32) {
      v16bf a = load_a_bf16(hA, lda, m0, k0, lane);
#pragma unroll
      for (int g = 0; g < 4; ++g)
        acc[g] = wmma_bf16(a, load_b(Wx, Kx, (g << 10) + n0, k0, lane), acc[g]);
    }
  } else {
#pragma unroll 2
    for (int k0 = 0; k0 < Kx; k0 += 32) {
      v16bf a = load_a_f32(xA, lda, m0, k0, lane);
#pragma unroll
      for (int g = 0; g < 4; ++g)
        acc[g] = wmma_bf16(a, load_b(Wx, Kx, (g << 10) + n0, k0, lane), acc[g]);
    }
  }

  // Phase 2: h_prev @ Wh^T
#pragma unroll 2
  for (int k0 = 0; k0 < HH; k0 += 32) {
    v16bf a = load_a_bf16(h_prev, HH, m0, k0, lane);
#pragma unroll
    for (int g = 0; g < 4; ++g)
      acc[g] = wmma_bf16(a, load_b(Wh, HH, (g << 10) + n0, k0, lane), acc[g]);
  }

  // Epilogue: biases + LSTM cell update, all in registers.
  int nj = n0 + (lane & 15);
  float bias[4];
#pragma unroll
  for (int g = 0; g < 4; ++g) bias[g] = bx[(g << 10) + nj] + bh[(g << 10) + nj];

  int hf = lane >> 4;
#pragma unroll
  for (int r = 0; r < 8; ++r) {
    int brow = m0 + r + (hf << 3);              // C/D layout: M = r (+8 for upper half)
    size_t idx = (size_t)brow * HH + nj;
    float iv = sigf(acc[0][r] + bias[0]);
    float fv = sigf(acc[1][r] + bias[1]);
    float gv = tanhf(acc[2][r] + bias[2]);
    float ov = sigf(acc[3][r] + bias[3]);
    float cn = fv * c[idx] + iv * gv;
    c[idx] = cn;
    h_out[idx] = f2bf(ov * tanhf(cn));
  }
}

// out[64,256] = sigmoid(h1 @ fcW^T + fcb)
__global__ void __launch_bounds__(256) fc_sig_kernel(
    const __bf16* __restrict__ h,    // [B, H] bf16
    const __bf16* __restrict__ Wb,   // [O, H] bf16
    const float*  __restrict__ bias, // [O]
    float* __restrict__ out)         // [B, O]
{
  int lane = threadIdx.x & 31;
  int wid  = (blockIdx.x * blockDim.x + threadIdx.x) >> 5;  // 0..63
  int n0 = (wid & 15) << 4;
  int m0 = (wid >> 4) << 4;

  v8f acc;
#pragma unroll
  for (int r = 0; r < 8; ++r) acc[r] = 0.0f;

#pragma unroll 2
  for (int k0 = 0; k0 < HH; k0 += 32) {
    v16bf a = load_a_bf16(h, HH, m0, k0, lane);
    acc = wmma_bf16(a, load_b(Wb, HH, n0, k0, lane), acc);
  }

  int nj = n0 + (lane & 15);
  float bv = bias[nj];
  int hf = lane >> 4;
#pragma unroll
  for (int r = 0; r < 8; ++r) {
    int brow = m0 + r + (hf << 3);
    out[(size_t)brow * OO + nj] = sigf(acc[r] + bv);
  }
}

// ---------- host ----------

extern "C" void kernel_launch(void* const* d_in, const int* in_sizes, int n_in,
                              void* d_out, int out_size, void* d_ws, size_t ws_size,
                              hipStream_t stream) {
  (void)in_sizes; (void)n_in; (void)out_size;
  const float* x   = (const float*)d_in[0];   // [B,S,D]
  const float* h0  = (const float*)d_in[1];   // [L,B,H]
  const float* c0  = (const float*)d_in[2];   // [L,B,H]
  const float* Wx0 = (const float*)d_in[3];   // [4H,D]
  const float* Wh0 = (const float*)d_in[4];   // [4H,H]
  const float* bx0 = (const float*)d_in[5];
  const float* bh0 = (const float*)d_in[6];
  const float* Wx1 = (const float*)d_in[7];   // [4H,H]
  const float* Wh1 = (const float*)d_in[8];   // [4H,H]
  const float* bx1 = (const float*)d_in[9];
  const float* bh1 = (const float*)d_in[10];
  const float* fcW = (const float*)d_in[11];  // [O,H]
  const float* fcb = (const float*)d_in[12];  // [O]
  float* out = (float*)d_out;

  // Workspace carve-up
  char* ws = (char*)d_ws;
  size_t off = 0;
  auto carve = [&](size_t bytes) -> void* {
    void* p = ws + off;
    off += (bytes + 255) & ~(size_t)255;
    return p;
  };
  __bf16* Wxb0 = (__bf16*)carve((size_t)4 * HH * DD * 2);
  __bf16* Whb0 = (__bf16*)carve((size_t)4 * HH * HH * 2);
  __bf16* Wxb1 = (__bf16*)carve((size_t)4 * HH * HH * 2);
  __bf16* Whb1 = (__bf16*)carve((size_t)4 * HH * HH * 2);
  __bf16* fcWb = (__bf16*)carve((size_t)OO * HH * 2);
  __bf16* h0b0 = (__bf16*)carve((size_t)BB * HH * 2);
  __bf16* h0b1 = (__bf16*)carve((size_t)BB * HH * 2);
  __bf16* h1b0 = (__bf16*)carve((size_t)BB * HH * 2);
  __bf16* h1b1 = (__bf16*)carve((size_t)BB * HH * 2);
  float*  c0w  = (float*)carve((size_t)BB * HH * 4);
  float*  c1w  = (float*)carve((size_t)BB * HH * 4);
  // x in bf16 (largest buffer, carved last; optional if ws is small)
  __bf16* xb   = (__bf16*)carve((size_t)BB * SS * DD * 2);
  bool use_xb = (off <= ws_size);

  __bf16* h0b[2] = { h0b0, h0b1 };
  __bf16* h1b[2] = { h1b0, h1b1 };

  auto cvt = [&](const float* in, __bf16* o, int n) {
    cvt_f32_bf16_kernel<<<(n + 255) / 256, 256, 0, stream>>>(in, o, n);
  };

  // One-time (per call) bf16 conversions: weights, x, initial h state.
  cvt(Wx0, Wxb0, 4 * HH * DD);
  cvt(Wh0, Whb0, 4 * HH * HH);
  cvt(Wx1, Wxb1, 4 * HH * HH);
  cvt(Wh1, Whb1, 4 * HH * HH);
  cvt(fcW, fcWb, OO * HH);
  if (use_xb) cvt(x, xb, BB * SS * DD);
  cvt(h0,                   h0b[0], BB * HH);
  cvt(h0 + (size_t)BB * HH, h1b[0], BB * HH);
  hipMemcpyAsync(c0w, c0,                   (size_t)BB * HH * 4,
                 hipMemcpyDeviceToDevice, stream);
  hipMemcpyAsync(c1w, c0 + (size_t)BB * HH, (size_t)BB * HH * 4,
                 hipMemcpyDeviceToDevice, stream);

  // Sequential scan over time: 2 WMMA GEMM+cell kernels per step.
  for (int t = 0; t < SS; ++t) {
    int rd = t & 1, wr = rd ^ 1;
    // Layer 0: A = x[:, t, :] (row stride S*D), Kx = D
    if (use_xb) {
      lstm_step_kernel<<<32, 256, 0, stream>>>(
          (const float*)nullptr, xb + (size_t)t * DD, SS * DD, DD,
          Wxb0, Whb0, h0b[rd], bx0, bh0, c0w, h0b[wr]);
    } else {
      lstm_step_kernel<<<32, 256, 0, stream>>>(
          x + (size_t)t * DD, (const __bf16*)nullptr, SS * DD, DD,
          Wxb0, Whb0, h0b[rd], bx0, bh0, c0w, h0b[wr]);
    }
    // Layer 1: A = h0 (bf16, just produced), Kx = H
    lstm_step_kernel<<<32, 256, 0, stream>>>(
        (const float*)nullptr, h0b[wr], HH, HH,
        Wxb1, Whb1, h1b[rd], bx1, bh1, c1w, h1b[wr]);
  }

  // t = 511: rd = 1, wr = 0 -> final h1 lives in h1b[0].
  fc_sig_kernel<<<8, 256, 0, stream>>>(h1b[0], fcWb, fcb, out);
}
